// MultiBackboneLateFusionRegressor_62517543961156
// MI455X (gfx1250) — matrix operand order repaired and checked by reading, not compiled, not run on hardware
//
#include <hip/hip_runtime.h>

typedef __attribute__((ext_vector_type(16))) __bf16 bf16x16;
typedef __attribute__((ext_vector_type(8)))  __bf16 bf16x8;
typedef __attribute__((ext_vector_type(8)))  float  f32x8;

#define H 64

// ---------------- utility kernels ----------------

__global__ void k_zero_f32(float* __restrict__ p, int n) {
    int i = blockIdx.x * blockDim.x + threadIdx.x;
    if (i < n) p[i] = 0.0f;
}

// Pack f32 row-major [64][64] weight matrices (K x N) into WMMA B-operand
// register order for V_WMMA_F32_16X16X32_BF16:
//   B tile (32x16): lane L holds column N=(L&15); element e holds K=(L>>4)*8+(e&7)+(e>>3)*16.
// Layout: [mat][ntile(4)][ktile(2)][lane(32)][e(16)]
__global__ void k_pack_w(const float* __restrict__ W, __bf16* __restrict__ P, int total) {
    int t = blockIdx.x * blockDim.x + threadIdx.x;
    if (t >= total) return;
    int m = t >> 12, r = t & 4095;
    int e    = r & 15;
    int lane = (r >> 4) & 31;
    int kt   = (r >> 9) & 1;
    int tn   = (r >> 10) & 3;
    int n = tn * 16 + (lane & 15);
    int k = kt * 32 + ((lane >> 4) << 3) + (e & 7) + ((e >> 3) << 4);
    P[t] = (__bf16)W[m * 4096 + k * 64 + n];
}

// input projection: pre = x @ w_in + b_in   (K=19, scalar FMA; trivially small)
__global__ void k_proj_in(const float* __restrict__ x, const float* __restrict__ w,
                          const float* __restrict__ b, float* __restrict__ pre,
                          int N, int Din) {
    int t = blockIdx.x * blockDim.x + threadIdx.x;
    if (t >= N * H) return;
    int i = t >> 6, c = t & 63;
    float acc = b[c];
    for (int k = 0; k < Din; ++k) acc = fmaf(x[i * Din + k], w[k * H + c], acc);
    pre[t] = acc;
}

// column-wise sum & sum-of-squares over node axis -> stats[0:64]=sum, [64:128]=sumsq
__global__ void k_col_stats(const float* __restrict__ pre, float* __restrict__ stats, int N) {
    int c   = threadIdx.x & 63;
    int rpb = blockDim.x >> 6;
    int i0  = blockIdx.x * rpb + (threadIdx.x >> 6);
    int str = gridDim.x * rpb;
    float s = 0.0f, s2 = 0.0f;
    for (int i = i0; i < N; i += str) {
        float v = pre[i * H + c];
        s += v;
        s2 = fmaf(v, v, s2);
    }
    atomicAdd(&stats[c], s);
    atomicAdd(&stats[H + c], s2);
}

// BN(train) + ReLU (+ optional residual); writes f32 h and bf16 copy for WMMA
__global__ void k_bn_finalize(const float* __restrict__ pre, const float* __restrict__ stats,
                              const float* __restrict__ g, const float* __restrict__ bb,
                              float* __restrict__ h, __bf16* __restrict__ hb,
                              float invN, int N, int residual) {
    int t = blockIdx.x * blockDim.x + threadIdx.x;
    if (t >= N * H) return;
    int c = t & 63;
    float mu  = stats[c] * invN;
    float var = stats[H + c] * invN - mu * mu;
    float v = (pre[t] - mu) * rsqrtf(var + 1e-5f) * g[c] + bb[c];
    v = fmaxf(v, 0.0f);
    if (residual) v += h[t];
    h[t]  = v;
    hb[t] = (__bf16)v;
}

// in-degree accumulation, then reciprocal of max(deg,1)
__global__ void k_deg(const int* __restrict__ dst, float* __restrict__ deg, int E) {
    int e = blockIdx.x * blockDim.x + threadIdx.x;
    if (e < E) atomicAdd(&deg[dst[e]], 1.0f);
}
__global__ void k_invdeg(float* __restrict__ deg, int N) {
    int i = blockIdx.x * blockDim.x + threadIdx.x;
    if (i < N) deg[i] = 1.0f / fmaxf(deg[i], 1.0f);
}

// edge scatter: agg[dst] += h[src]  (64 threads per edge, f32 atomics into L2)
__global__ void k_scatter(const int* __restrict__ src, const int* __restrict__ dst,
                          const float* __restrict__ h, float* __restrict__ agg, int E) {
    int t = blockIdx.x * blockDim.x + threadIdx.x;
    int e = t >> 6;
    if (e >= E) return;
    int c = t & 63;
    atomicAdd(&agg[dst[e] * H + c], h[src[e] * H + c]);
}

// mean-scale + bf16 convert
__global__ void k_agg_scale(const float* __restrict__ agg, const float* __restrict__ inv,
                            __bf16* __restrict__ aggb, int N) {
    int t = blockIdx.x * blockDim.x + threadIdx.x;
    if (t >= N * H) return;
    aggb[t] = (__bf16)(agg[t] * inv[t >> 6]);
}

// ---------------- WMMA GEMM: pre = aggb@Wl + hb@Wr + bias ----------------
// One wave per 16x16 output tile (flattened (mtile, ntile) index) so the
// per-wave register set (A: 32 VGPRs, B: 32 VGPRs, acc: 8) fits with no
// scratch spills.
// A fragment (16x32 bf16): lanes 0-15 row M=lane, elems = K klo..klo+7,
// klo+16..klo+23 with klo=(lane>>4)*8.  B pre-packed in register order.
// C/D: elem i -> row M = i + (lane>>4)*8, col N = lane&15.
__global__ void __launch_bounds__(256)
k_gemm_wmma(const __bf16* __restrict__ aggb, const __bf16* __restrict__ hb,
            const __bf16* __restrict__ pWl, const __bf16* __restrict__ pWr,
            const float* __restrict__ bias, float* __restrict__ pre,
            int mtiles) {
    const int lane = threadIdx.x & 31;
    const int wave = threadIdx.x >> 5;
    const int wpb  = blockDim.x >> 5;
    const int r    = lane & 15;
    const int klo  = (lane >> 4) * 8;
    const int ntot = mtiles * 4;

    for (int idx = blockIdx.x * wpb + wave; idx < ntot; idx += gridDim.x * wpb) {
        const int mt = idx >> 2;
        const int tn = idx & 3;
        const int row = mt * 16 + r;

        bf16x16 a_h[2], a_g[2];
#pragma unroll
        for (int kt = 0; kt < 2; ++kt) {
            const __bf16* ph = hb   + row * H + kt * 32 + klo;
            const __bf16* pg = aggb + row * H + kt * 32 + klo;
            bf16x8 hlo = *(const bf16x8*)(ph);
            bf16x8 hhi = *(const bf16x8*)(ph + 16);
            bf16x8 glo = *(const bf16x8*)(pg);
            bf16x8 ghi = *(const bf16x8*)(pg + 16);
#pragma unroll
            for (int i = 0; i < 8; ++i) {
                a_h[kt][i] = hlo[i]; a_h[kt][8 + i] = hhi[i];
                a_g[kt][i] = glo[i]; a_g[kt][8 + i] = ghi[i];
            }
        }

        const int c = tn * 16 + r;
        float bv = bias[c];
        f32x8 acc;
#pragma unroll
        for (int i = 0; i < 8; ++i) acc[i] = bv;

#pragma unroll
        for (int kt = 0; kt < 2; ++kt) {
            bf16x16 bL = *(const bf16x16*)(pWl + ((tn * 2 + kt) * 32 + lane) * 16);
            bf16x16 bR = *(const bf16x16*)(pWr + ((tn * 2 + kt) * 32 + lane) * 16);
            acc = __builtin_amdgcn_wmma_f32_16x16x32_bf16(
                      false, a_g[kt], false, bL, (short)0, acc, false, false);
            acc = __builtin_amdgcn_wmma_f32_16x16x32_bf16(
                      false, a_h[kt], false, bR, (short)0, acc, false, false);
        }

        const int mbase = mt * 16 + ((lane >> 4) << 3);
#pragma unroll
        for (int i = 0; i < 8; ++i) pre[(mbase + i) * H + c] = acc[i];
    }
}

// scalar tail for rows beyond the last full 16-row tile (N=50000 -> unused)
__global__ void k_gemm_tail(const __bf16* __restrict__ aggb, const __bf16* __restrict__ hb,
                            const float* __restrict__ wlm, const float* __restrict__ wrm,
                            const float* __restrict__ bias, float* __restrict__ pre,
                            int rowStart, int N) {
    int t = blockIdx.x * blockDim.x + threadIdx.x;
    int nrows = N - rowStart;
    if (t >= nrows * H) return;
    int i = rowStart + (t >> 6), c = t & 63;
    float acc = bias[c];
    for (int k = 0; k < H; ++k)
        acc += (float)aggb[i * H + k] * wlm[k * H + c] + (float)hb[i * H + k] * wrm[k * H + c];
    pre[i * H + c] = acc;
}

// ---------------- launcher ----------------

extern "C" void kernel_launch(void* const* d_in, const int* in_sizes, int n_in,
                              void* d_out, int out_size, void* d_ws, size_t ws_size,
                              hipStream_t stream) {
    (void)n_in; (void)out_size; (void)ws_size;

    const float* x    = (const float*)d_in[0];
    const int*   ei   = (const int*)d_in[1];
    const float* w_in = (const float*)d_in[2];
    const float* b_in = (const float*)d_in[3];
    const float* bn0g = (const float*)d_in[4];
    const float* bn0b = (const float*)d_in[5];
    const float* wl   = (const float*)d_in[6];
    const float* bl   = (const float*)d_in[7];
    const float* wr   = (const float*)d_in[8];
    const float* bng  = (const float*)d_in[9];
    const float* bnb  = (const float*)d_in[10];

    const int Din = 19, L = 3;
    const int N = in_sizes[0] / Din;
    const int E = in_sizes[1] / 2;
    const int* src  = ei;       // edge_index[0]
    const int* dstp = ei + E;   // edge_index[1]

    float* h = (float*)d_out;   // evolving node features live in d_out

    // workspace carving (256B aligned)
    char* p = (char*)d_ws;
    auto carve = [&](size_t bytes) -> char* {
        char* r = p;
        p += (bytes + 255) & ~(size_t)255;
        return r;
    };
    __bf16* hb    = (__bf16*)carve((size_t)N * H * 2);
    __bf16* aggb  = (__bf16*)carve((size_t)N * H * 2);
    float*  agg   = (float*)carve((size_t)N * H * 4);
    float*  pre   = (float*)carve((size_t)N * H * 4);
    float*  invd  = (float*)carve((size_t)N * 4);
    float*  stats = (float*)carve(2 * H * 4);
    __bf16* pWl   = (__bf16*)carve((size_t)L * 4096 * 2);
    __bf16* pWr   = (__bf16*)carve((size_t)L * 4096 * 2);

    const int T = 256;
    const int NH = N * H;
    const float invN = 1.0f / (float)N;
    const int mtiles = N / 16;
    const int tail = N - mtiles * 16;
    const int gemmTiles = mtiles * 4;               // (mtile, ntile) pairs
    const int gemmBlocks = (gemmTiles + 7) / 8;     // 8 waves per 256-thread block
    const unsigned scatterBlocks = (unsigned)(((long long)E * H + T - 1) / T);

    // pack weights into WMMA B-operand order
    k_pack_w<<<(L * 4096 + T - 1) / T, T, 0, stream>>>(wl, pWl, L * 4096);
    k_pack_w<<<(L * 4096 + T - 1) / T, T, 0, stream>>>(wr, pWr, L * 4096);

    // degree -> 1/max(deg,1)
    k_zero_f32<<<(N + T - 1) / T, T, 0, stream>>>(invd, N);
    k_deg<<<(E + T - 1) / T, T, 0, stream>>>(dstp, invd, E);
    k_invdeg<<<(N + T - 1) / T, T, 0, stream>>>(invd, N);

    // input projection -> BN -> ReLU
    k_proj_in<<<(NH + T - 1) / T, T, 0, stream>>>(x, w_in, b_in, pre, N, Din);
    k_zero_f32<<<1, 2 * H, 0, stream>>>(stats, 2 * H);
    k_col_stats<<<128, T, 0, stream>>>(pre, stats, N);
    k_bn_finalize<<<(NH + T - 1) / T, T, 0, stream>>>(pre, stats, bn0g, bn0b,
                                                      h, hb, invN, N, 0);

    for (int l = 0; l < L; ++l) {
        k_zero_f32<<<(NH + T - 1) / T, T, 0, stream>>>(agg, NH);
        k_scatter<<<scatterBlocks, T, 0, stream>>>(src, dstp, h, agg, E);
        k_agg_scale<<<(NH + T - 1) / T, T, 0, stream>>>(agg, invd, aggb, N);

        k_gemm_wmma<<<gemmBlocks, 256, 0, stream>>>(
            aggb, hb, pWl + (size_t)l * 4096, pWr + (size_t)l * 4096,
            bl + l * H, pre, mtiles);
        if (tail > 0)
            k_gemm_tail<<<(tail * H + T - 1) / T, T, 0, stream>>>(
                aggb, hb, wl + (size_t)l * 4096, wr + (size_t)l * 4096,
                bl + l * H, pre, mtiles * 16, N);

        k_zero_f32<<<1, 2 * H, 0, stream>>>(stats, 2 * H);
        k_col_stats<<<128, T, 0, stream>>>(pre, stats, N);
        k_bn_finalize<<<(NH + T - 1) / T, T, 0, stream>>>(
            pre, stats, bng + l * H, bnb + l * H, h, hb, invN, N, 1);
    }
}
